// GCN_64123861729597
// MI455X (gfx1250) — matrix operand order repaired and checked by reading, not compiled, not run on hardware
//
#include <hip/hip_runtime.h>
#include <hip/hip_bf16.h>
#include <math.h>

// ---------------------------------------------------------------------------
// CDNA5 (gfx1250) GCN + bilinear pooling pipeline.
// Dense GEMMs: v_wmma_f32_16x16x32_bf16 (wave32), weights pre-packed to bf16
// in WMMA B-fragment order; fragments preloaded into distinct registers, then
// WMMA bursts (no per-iteration waits, no hazard NOPs, immediate offsets).
// ---------------------------------------------------------------------------

typedef __attribute__((ext_vector_type(16))) __bf16 v16bf;
typedef __attribute__((ext_vector_type(8)))  float  v8f;

union BF16Frag {
    v16bf          v;
    unsigned short h[16];
    uint4          q[2];
};

// round-to-nearest-even float -> bf16 (bit pattern as ushort)
__device__ __forceinline__ unsigned short f2bf(float f) {
    union { float f; unsigned u; } x;
    x.f = f;
    unsigned r = x.u + 0x7FFFu + ((x.u >> 16) & 1u);
    return (unsigned short)(r >> 16);
}

__device__ __forceinline__ unsigned pack2bf(float lo, float hi) {
    return (unsigned)f2bf(lo) | ((unsigned)f2bf(hi) << 16);
}

// ---------------------------------------------------------------------------
// A-fragment (16x32 bf16, MxK) from LDS, row-major staging [16][ld] halfwords.
// ISA 7.12.2: lanes 0-15: row=lane, elems 0..7 -> K=kbase+0..7, 8..15 -> +16..23
//             lanes16-31: row=lane-16, halves shifted by 8.
// ---------------------------------------------------------------------------
__device__ __forceinline__ v16bf load_a_lds(const unsigned short* smem, int ld, int kbase) {
    const int lane  = threadIdx.x & 31;
    const int row   = lane & 15;
    const int khalf = (lane >> 4) * 8;
    BF16Frag a;
    const unsigned short* p = smem + row * ld + kbase + khalf;
    a.q[0] = *reinterpret_cast<const uint4*>(p);
    a.q[1] = *reinterpret_cast<const uint4*>(p + 16);
    return a.v;
}

// B-fragment from packed weights: 32 bytes contiguous per lane -> 2x b128.
__device__ __forceinline__ v16bf load_b_packed(const unsigned short* __restrict__ pB,
                                               size_t halfword_off) {
    const uint4* p = reinterpret_cast<const uint4*>(pB + halfword_off);
    BF16Frag b;
    b.q[0] = p[0];
    b.q[1] = p[1];
    return b.v;
}

__device__ __forceinline__ v8f wmma_bf16(v16bf a, v16bf b, v8f c) {
    return __builtin_amdgcn_wmma_f32_16x16x32_bf16(false, a, false, b, (short)0, c,
                                                   false, false);
}

// ---------------------------------------------------------------------------
// Pack fp32 weight W[K][Nout] into bf16 WMMA B-fragment order.
// Per (ktile,ntile) 32x16 tile: lane l: col=l&15, k=kt*32+(l>>4)*8+(e&7)+(e>>3)*16.
// ---------------------------------------------------------------------------
__global__ void pack_weight_kernel(const float* __restrict__ W,
                                   unsigned short* __restrict__ Wb, int K, int Nout) {
    const int ntiles = Nout >> 4;
    const int total  = K * Nout;
    for (int o = blockIdx.x * blockDim.x + threadIdx.x; o < total;
         o += gridDim.x * blockDim.x) {
        const int frag   = o >> 9;
        const int within = o & 511;
        const int lane   = within >> 4;
        const int e      = within & 15;
        const int ktile  = frag / ntiles;
        const int ntile  = frag - ktile * ntiles;
        const int col    = lane & 15;
        const int k      = ktile * 32 + ((lane >> 4) << 3) + (e & 7) + ((e >> 3) << 4);
        Wb[o] = f2bf(W[k * Nout + ntile * 16 + col]);
    }
}

// ---------------------------------------------------------------------------
// GEMM + optional bias: C[M][NOUT] = A[M][K] @ W (+bias), packed Wb.
// 32 rows per block (two 16-row accumulators per wave, shared B fragments).
// block = 256 = 8 waves; wave w -> column tile w.  K in {128,256}, NOUT%16==0.
// ---------------------------------------------------------------------------
template <int K, int NOUT>
__global__ void __launch_bounds__(256)
gemm_bias_kernel(const float* __restrict__ A, const unsigned short* __restrict__ Wb,
                 const float* __restrict__ bias, float* __restrict__ C, int M) {
    constexpr int NT    = NOUT >> 4;
    constexpr int MROWS = 32;
    __shared__ unsigned int sAu[MROWS * K / 2];

    const int mbase = blockIdx.x * MROWS;

    // flat bf16 staging of 32 contiguous rows; zero-pad past M (tail block)
    {
        const float2* Arows = reinterpret_cast<const float2*>(A + (size_t)mbase * K);
        const int npairs = (MROWS * K) >> 1;
        const int nvalid = ((M - mbase) >= MROWS) ? npairs : (((M - mbase) * K) >> 1);
        for (int i = threadIdx.x; i < npairs; i += 256) {
            float2 f = (i < nvalid) ? Arows[i] : float2{0.0f, 0.0f};
            sAu[i] = pack2bf(f.x, f.y);
        }
    }
    __syncthreads();

    const int wave  = threadIdx.x >> 5;
    const int lane  = threadIdx.x & 31;
    const int ntile = wave;
    if (ntile >= NT) return;                  // wave-uniform exit (NOUT==64 case)

    const unsigned short* sA = reinterpret_cast<const unsigned short*>(sAu);
    const unsigned short* pB = Wb + (((size_t)ntile * 32 + lane) << 4);

    v8f acc0 = {}, acc1 = {};
#pragma unroll
    for (int kc = 0; kc < K; kc += 128) {
        v16bf a0[4], a1[4], bf[4];
#pragma unroll
        for (int kt = 0; kt < 4; ++kt) {      // preload into distinct registers
            a0[kt] = load_a_lds(sA,           K, kc + kt * 32);
            a1[kt] = load_a_lds(sA + 16 * K,  K, kc + kt * 32);
            bf[kt] = load_b_packed(pB, (size_t)((kc >> 5) + kt) * NT * 512);
        }
#pragma unroll
        for (int kt = 0; kt < 4; ++kt) {      // WMMA burst, B reused twice
            acc0 = wmma_bf16(a0[kt], bf[kt], acc0);
            acc1 = wmma_bf16(a1[kt], bf[kt], acc1);
        }
    }

    const int col  = ntile * 16 + (lane & 15);
    const float bv = bias ? bias[col] : 0.0f;
    const int r0   = mbase + (lane >> 4) * 8;
#pragma unroll
    for (int v = 0; v < 8; ++v) {
        const int ra = r0 + v, rb = r0 + v + 16;
        if (ra < M) C[(size_t)ra * NOUT + col] = acc0[v] + bv;
        if (rb < M) C[(size_t)rb * NOUT + col] = acc1[v] + bv;
    }
}

// ---------------------------------------------------------------------------
// Fused bilinear pooling (16 rows per block):
//   R = (Hn@Wn + bn) * (Hm@Wm + bm)   [16 x 640] in LDS
//   iq[:,j] = sum_{g<5} R[:,5j+g]; iq = sign(iq)*sqrt(|iq|); L2-normalize rows.
// A fragments hoisted out of the column-tile loop (invariant).
// ---------------------------------------------------------------------------
__global__ void __launch_bounds__(256)
bilinear_pool_kernel(const float* __restrict__ Hn, const float* __restrict__ Hm,
                     const unsigned short* __restrict__ Wn, const float* __restrict__ bn,
                     const unsigned short* __restrict__ Wm, const float* __restrict__ bm,
                     float* __restrict__ Out) {
    __shared__ unsigned int sAu[2 * 16 * 128 / 2];  // node rows | neighbor rows (bf16)
    __shared__ float sR[16 * 640];                  // 40 KB product buffer
    __shared__ float sIQ[16 * 128];
    __shared__ float sNorm[16 * 16];

    const int mbase = blockIdx.x * 16;

    const float2* HnRows = reinterpret_cast<const float2*>(Hn + (size_t)mbase * 128);
    const float2* HmRows = reinterpret_cast<const float2*>(Hm + (size_t)mbase * 128);
    for (int i = threadIdx.x; i < 16 * 128 / 2; i += 256) {
        const float2 a = HnRows[i];
        const float2 b = HmRows[i];
        sAu[i]                = pack2bf(a.x, a.y);
        sAu[16 * 128 / 2 + i] = pack2bf(b.x, b.y);
    }
    __syncthreads();

    const int wave = threadIdx.x >> 5;
    const int lane = threadIdx.x & 31;
    const unsigned short* sA = reinterpret_cast<const unsigned short*>(sAu);

    // invariant A fragments: 4 node + 4 neighbor
    v16bf an[4], am[4];
#pragma unroll
    for (int kt = 0; kt < 4; ++kt) {
        an[kt] = load_a_lds(sA,            128, kt * 32);
        am[kt] = load_a_lds(sA + 16 * 128, 128, kt * 32);
    }

    // 40 column tiles of the 640-wide projections, 5 per wave
    for (int t = wave; t < 40; t += 8) {
        const size_t tilebase = ((size_t)t * 32 + lane) << 4;
        v16bf bnf[4], bmf[4];
#pragma unroll
        for (int kt = 0; kt < 4; ++kt) {      // preload B fragments
            bnf[kt] = load_b_packed(Wn + tilebase, (size_t)kt * 40 * 512);
            bmf[kt] = load_b_packed(Wm + tilebase, (size_t)kt * 40 * 512);
        }
        v8f p = {}, q = {};
#pragma unroll
        for (int kt = 0; kt < 4; ++kt) {      // 8-WMMA burst
            p = wmma_bf16(an[kt], bnf[kt], p);
            q = wmma_bf16(am[kt], bmf[kt], q);
        }
        const int col    = t * 16 + (lane & 15);
        const float bn_v = bn[col];
        const float bm_v = bm[col];
        const int rb     = (lane >> 4) * 8;
#pragma unroll
        for (int v = 0; v < 8; ++v)
            sR[(rb + v) * 640 + col] = (p[v] + bn_v) * (q[v] + bm_v);
    }
    __syncthreads();

    // group-of-5 sum + signed sqrt; 16 threads per row, 8 feats per thread
    const int r  = threadIdx.x >> 4;
    const int c0 = (threadIdx.x & 15) * 8;
    float ss = 0.0f;
#pragma unroll
    for (int j = 0; j < 8; ++j) {
        const int c = c0 + j;
        float s = 0.0f;
#pragma unroll
        for (int g = 0; g < 5; ++g) s += sR[r * 640 + c * 5 + g];
        const float v = (s >= 0.0f) ? sqrtf(s) : -sqrtf(-s);
        sIQ[r * 128 + c] = v;
        ss += v * v;
    }
    sNorm[r * 16 + (threadIdx.x & 15)] = ss;
    __syncthreads();

    if (threadIdx.x < 16) {
        float tot = 0.0f;
#pragma unroll
        for (int i = 0; i < 16; ++i) tot += sNorm[threadIdx.x * 16 + i];
        sNorm[threadIdx.x * 16] = fmaxf(sqrtf(tot), 1e-12f);
    }
    __syncthreads();

    const float inv = 1.0f / sNorm[r * 16];
#pragma unroll
    for (int j = 0; j < 8; ++j) {
        const int c = c0 + j;
        Out[(size_t)(mbase + r) * 128 + c] = sIQ[r * 128 + c] * inv;
    }
}

// ---------------------------------------------------------------------------
// Out[i][f] = b[f]  (SpMM bias pre-init)
// ---------------------------------------------------------------------------
__global__ void rowbias_init_kernel(const float* __restrict__ b, float* __restrict__ Out,
                                    int total, int F) {
    const int gid = blockIdx.x * blockDim.x + threadIdx.x;
    if (gid < total) Out[gid] = b[gid & (F - 1)];   // F is a power of two
}

// ---------------------------------------------------------------------------
// SpMM scatter: Out[src[e]] += val[e] * T[dst[e]]   (one wave per edge)
// ---------------------------------------------------------------------------
__global__ void spmm_kernel(const int* __restrict__ src, const int* __restrict__ dst,
                            const float* __restrict__ val, const float* __restrict__ T,
                            float* __restrict__ Out, int nedge, int F) {
    const int gid  = blockIdx.x * blockDim.x + threadIdx.x;
    const int e    = gid >> 5;
    const int lane = gid & 31;
    if (e >= nedge) return;
    const int   s = src[e];
    const int   d = dst[e];
    const float v = val[e];
    for (int f = lane; f < F; f += 32)
        atomicAdd(&Out[(size_t)s * F + f], v * T[(size_t)d * F + f]);
}

// ---------------------------------------------------------------------------
// In-place log_softmax over 64 classes; one wave per row (2 cols / lane).
// ---------------------------------------------------------------------------
__global__ void __launch_bounds__(256)
logsoftmax_kernel(float* __restrict__ X, int n) {
    const int row  = blockIdx.x * 8 + (threadIdx.x >> 5);
    const int lane = threadIdx.x & 31;
    if (row >= n) return;
    float v0 = X[row * 64 + lane];
    float v1 = X[row * 64 + lane + 32];
    float m = fmaxf(v0, v1);
#pragma unroll
    for (int off = 16; off > 0; off >>= 1)
        m = fmaxf(m, __shfl_xor(m, off, 32));
    float s = __expf(v0 - m) + __expf(v1 - m);
#pragma unroll
    for (int off = 16; off > 0; off >>= 1)
        s += __shfl_xor(s, off, 32);
    const float l = logf(s);
    X[row * 64 + lane]      = v0 - m - l;
    X[row * 64 + lane + 32] = v1 - m - l;
}

// ---------------------------------------------------------------------------
// Host-side orchestration
// ---------------------------------------------------------------------------
extern "C" void kernel_launch(void* const* d_in, const int* in_sizes, int n_in,
                              void* d_out, int out_size, void* d_ws, size_t ws_size,
                              hipStream_t stream) {
    (void)in_sizes; (void)n_in; (void)out_size; (void)ws_size;

    constexpr int N      = 50000;
    constexpr int NFEAT  = 256;
    constexpr int NHID   = 128;
    constexpr int NCLASS = 64;
    constexpr int NEDGE  = 800000;

    const float* x        = (const float*)d_in[0];
    const int*   edge_src = (const int*)  d_in[1];
    const int*   edge_dst = (const int*)  d_in[2];
    const float* edge_val = (const float*)d_in[3];
    const float* W_enc    = (const float*)d_in[4];
    const float* b_enc    = (const float*)d_in[5];
    const float* W_gc1    = (const float*)d_in[6];
    const float* b_gc1    = (const float*)d_in[7];
    const float* W_gc2    = (const float*)d_in[8];
    const float* b_gc2    = (const float*)d_in[9];
    const float* W_gc3    = (const float*)d_in[10];
    const float* b_gc3    = (const float*)d_in[11];
    const float* W_node   = (const float*)d_in[12];
    const float* b_node   = (const float*)d_in[13];
    const float* W_nei    = (const float*)d_in[14];
    const float* b_nei    = (const float*)d_in[15];
    float* out = (float*)d_out;

    // ---- workspace layout: 4 x [N,128] f32, then bf16 packed weights -------
    float* x_org = (float*)d_ws;                 // [N,128]
    float* t     = x_org + (size_t)N * NHID;     // [N,128] (also [N,64] for gc3)
    float* h     = t     + (size_t)N * NHID;     // [N,128]
    float* x1    = h     + (size_t)N * NHID;     // [N,128]
    float* x2    = x_org;                        // reuse once x_org is dead

    unsigned short* Wb_enc  = (unsigned short*)(x1 + (size_t)N * NHID);
    unsigned short* Wb_gc1  = Wb_enc  + NFEAT * NHID;        // 256*128
    unsigned short* Wb_gc2  = Wb_gc1  + NHID * NHID;         // 128*128
    unsigned short* Wb_gc3  = Wb_gc2  + NHID * NHID;         // 128*128
    unsigned short* Wb_node = Wb_gc3  + NHID * NCLASS;       // 128*64
    unsigned short* Wb_nei  = Wb_node + NHID * 5 * NHID;     // 128*640

    const dim3 blk(256);
    const int  mtiles16   = N / 16;                          // 3125 (pool)
    const int  mtiles32   = (N + 31) / 32;                   // 1563 (gemm)
    const int  spmmBlk    = (NEDGE * 32 + 255) / 256;
    const int  biasBlk128 = (N * NHID + 255) / 256;
    const int  biasBlk64  = (N * NCLASS + 255) / 256;

    // ---- pack all weights to bf16 fragment order (once, tiny) --------------
    pack_weight_kernel<<<(NFEAT * NHID + 255) / 256, blk, 0, stream>>>(W_enc, Wb_enc, NFEAT, NHID);
    pack_weight_kernel<<<(NHID * NHID + 255) / 256, blk, 0, stream>>>(W_gc1, Wb_gc1, NHID, NHID);
    pack_weight_kernel<<<(NHID * NHID + 255) / 256, blk, 0, stream>>>(W_gc2, Wb_gc2, NHID, NHID);
    pack_weight_kernel<<<(NHID * NCLASS + 255) / 256, blk, 0, stream>>>(W_gc3, Wb_gc3, NHID, NCLASS);
    pack_weight_kernel<<<(NHID * 5 * NHID + 255) / 256, blk, 0, stream>>>(W_node, Wb_node, NHID, 5 * NHID);
    pack_weight_kernel<<<(NHID * 5 * NHID + 255) / 256, blk, 0, stream>>>(W_nei, Wb_nei, NHID, 5 * NHID);

    // ---- pipeline -----------------------------------------------------------
    // encoder: x_org = x @ W_enc + b_enc
    gemm_bias_kernel<256, 128><<<mtiles32, blk, 0, stream>>>(x, Wb_enc, b_enc, x_org, N);
    // gc1: h = spmm(x_org @ W_gc1) + b_gc1
    gemm_bias_kernel<128, 128><<<mtiles32, blk, 0, stream>>>(x_org, Wb_gc1, nullptr, t, N);
    rowbias_init_kernel<<<biasBlk128, blk, 0, stream>>>(b_gc1, h, N * NHID, NHID);
    spmm_kernel<<<spmmBlk, blk, 0, stream>>>(edge_src, edge_dst, edge_val, t, h,
                                             NEDGE, NHID);
    // pool 1: x1 = bilinear(x_org, h)
    bilinear_pool_kernel<<<mtiles16, blk, 0, stream>>>(x_org, h, Wb_node, b_node,
                                                       Wb_nei, b_nei, x1);
    // gc2
    gemm_bias_kernel<128, 128><<<mtiles32, blk, 0, stream>>>(x1, Wb_gc2, nullptr, t, N);
    rowbias_init_kernel<<<biasBlk128, blk, 0, stream>>>(b_gc2, h, N * NHID, NHID);
    spmm_kernel<<<spmmBlk, blk, 0, stream>>>(edge_src, edge_dst, edge_val, t, h,
                                             NEDGE, NHID);
    // pool 2: x2 = bilinear(x1, h)
    bilinear_pool_kernel<<<mtiles16, blk, 0, stream>>>(x1, h, Wb_node, b_node,
                                                       Wb_nei, b_nei, x2);
    // gc3: out = spmm(x2 @ W_gc3) + b_gc3
    gemm_bias_kernel<128, 64><<<mtiles32, blk, 0, stream>>>(x2, Wb_gc3, nullptr, t, N);
    rowbias_init_kernel<<<biasBlk64, blk, 0, stream>>>(b_gc3, out, N * NCLASS, NCLASS);
    spmm_kernel<<<spmmBlk, blk, 0, stream>>>(edge_src, edge_dst, edge_val, t, out,
                                             NEDGE, NCLASS);
    // log_softmax rows of 64
    logsoftmax_kernel<<<(N + 7) / 8, blk, 0, stream>>>(out, N);
}
